// AffineLayer_24197845745735
// MI455X (gfx1250) — matrix operand order repaired and checked by reading, not compiled, not run on hardware
//
#include <hip/hip_runtime.h>
#include <hip/hip_bf16.h>

// ---------------------------------------------------------------------------
// Fused  out[b,f] = max_p( x[b,:] . ww[f,p,:] ) * scale[f] + bias[f]
// B=2048, F=1024, P=64, K=i=64.  Compute-bound (~690 FLOP/byte) -> bf16 WMMA
// (v_wmma_f32_16x16x32_bf16), f32 accumulation over K=64.
//
// Transposed tiling: A = ww tile (M=p, K=i), B = x tile (K=i, N=b), so the
// max over p reduces over VGPR components + one lane-half swap instead of a
// 16-lane butterfly per component.  global_prefetch warms the next f's 8KB
// ww region in the WGP cache while the current f's WMMAs run (prefetches
// can't be register-coalesced away, unlike early loads).
// ---------------------------------------------------------------------------

typedef __bf16 v16bf __attribute__((ext_vector_type(16)));
typedef __bf16 bf4v  __attribute__((ext_vector_type(4)));
typedef float  v8f   __attribute__((ext_vector_type(8)));

constexpr int kB = 2048;   // batch
constexpr int kF = 1024;   // features (output dim)
constexpr int kP = 64;     // samples (max-pooled)
constexpr int kK = 64;     // inner dim C*H*W

// ----------------------------- f32 -> bf16 pre-pass -------------------------
__global__ __launch_bounds__(256) void cvt_f32_to_bf16(
    const float* __restrict__ src, __bf16* __restrict__ dst, int n4) {
  int i = blockIdx.x * blockDim.x + threadIdx.x;
  if (i < n4) {
    float4 f = reinterpret_cast<const float4*>(src)[i];
    bf4v o = { (__bf16)f.x, (__bf16)f.y, (__bf16)f.z, (__bf16)f.w };
    reinterpret_cast<bf4v*>(dst)[i] = o;
  }
}

// ----------------------------- fused GEMM + max-pool ------------------------
// grid = (B/16, F/32), block = 256 threads (8 waves).
// wave -> 16 b-columns x 4 f.  Per f: M = p in 4 tiles of 16, K = 64.
__global__ __launch_bounds__(256) void gemm_maxpool_affine(
    const __bf16* __restrict__ xb,      // (B, K) bf16, row-major
    const __bf16* __restrict__ wb,      // (F, P, K) bf16, K contiguous
    const float*  __restrict__ scale,   // (F)
    const float*  __restrict__ bias,    // (F)
    float*        __restrict__ out)     // (B, F) f32
{
  const int lane = threadIdx.x & 31;
  const int wave = threadIdx.x >> 5;
  const int bt   = blockIdx.x;          // 16-column b tile
  const int fg   = blockIdx.y;          // 32-wide f group
  const int ln   = lane & 15;           // N index (b) / M index within A rows
  const int kh   = lane >> 4;           // K-half selector per ISA layouts

  union Frag { uint4 q[2]; v16bf v; };

  const int f0 = fg * 32 + wave * 4;

  // Warm the WGP cache for the first f: 32 lanes x 64B lines x 4 tiles = 8KB
  {
    const __bf16* wp0 = wb + (size_t)f0 * (kP * kK) + (size_t)ln * kK + kh * 32;
    #pragma unroll
    for (int t = 0; t < 4; ++t)
      __builtin_prefetch(wp0 + t * 16 * kK, 0, 3);
  }

  // B operand: x columns.  B 32x16 layout: lane = column N (b = bt*16+ln),
  // halves[0..7] = K = kh*8+(0..7), halves[8..15] = K = 16+kh*8+(0..7) (matches A).
  const __bf16* xcol = xb + (size_t)(bt * 16 + ln) * kK + kh * 8;
  Frag xb0u, xb1u;
  xb0u.q[0] = *reinterpret_cast<const uint4*>(xcol);          // K 0..31
  xb0u.q[1] = *reinterpret_cast<const uint4*>(xcol + 16);
  xb1u.q[0] = *reinterpret_cast<const uint4*>(xcol + 32);     // K 32..63
  xb1u.q[1] = *reinterpret_cast<const uint4*>(xcol + 48);
  const v16bf xB0 = xb0u.v, xB1 = xb1u.v;

  const float4 sc4 = *reinterpret_cast<const float4*>(scale + f0);
  const float4 bi4 = *reinterpret_cast<const float4*>(bias + f0);

  float res[4];

  #pragma unroll
  for (int fi = 0; fi < 4; ++fi) {
    const int f = f0 + fi;
    const __bf16* wf = wb + (size_t)f * (kP * kK);

    // Prefetch the NEXT f's whole 8KB region (4 instructions, no VGPR cost)
    if (fi < 3) {
      const __bf16* wn = wf + (kP * kK) + (size_t)ln * kK + kh * 32;
      #pragma unroll
      for (int t = 0; t < 4; ++t)
        __builtin_prefetch(wn + t * 16 * kK, 0, 3);
    }

    v8f acc[4];
    #pragma unroll
    for (int t = 0; t < 4; ++t) {
      // A operand: ww rows (M = p = t*16 + ln).  16-bit A 16x32 layout:
      // halves[0..7] = K = kh*8+(0..7), halves[8..15] = K = 16+kh*8+(0..7).
      const __bf16* wp = wf + (size_t)(t * 16 + ln) * kK;
      Frag a0u, a1u;
      a0u.q[0] = *reinterpret_cast<const uint4*>(wp + kh * 8);
      a0u.q[1] = *reinterpret_cast<const uint4*>(wp + 16 + kh * 8);
      a1u.q[0] = *reinterpret_cast<const uint4*>(wp + 32 + kh * 8);
      a1u.q[1] = *reinterpret_cast<const uint4*>(wp + 48 + kh * 8);

      v8f c = {};
      c = __builtin_amdgcn_wmma_f32_16x16x32_bf16(
              false, a0u.v, false, xB0, (short)0, c, false, false);   // K 0..31
      c = __builtin_amdgcn_wmma_f32_16x16x32_bf16(
              false, a1u.v, false, xB1, (short)0, c, false, false);   // K 32..63
      acc[t] = c;
    }

    // p sits on (tile, component, lane-half): component c of lane-half g in
    // tile t is p = t*16 + c + 8*g, column b = bt*16 + ln for both halves.
    float m = fmaxf(fmaxf(acc[0][0], acc[1][0]), fmaxf(acc[2][0], acc[3][0]));
    #pragma unroll
    for (int c = 1; c < 8; ++c) {
      float mc = fmaxf(fmaxf(acc[0][c], acc[1][c]), fmaxf(acc[2][c], acc[3][c]));
      m = fmaxf(m, mc);
    }
    // merge the two disjoint p-halves held by lane and lane^16
    m = fmaxf(m, __shfl_xor(m, 16, 32));
    res[fi] = m;
  }

  // lanes 0-15 each own one b row; 4 consecutive f -> one 16B store per lane
  if (lane < 16) {
    float4 o;
    o.x = res[0] * sc4.x + bi4.x;
    o.y = res[1] * sc4.y + bi4.y;
    o.z = res[2] * sc4.z + bi4.z;
    o.w = res[3] * sc4.w + bi4.w;
    *reinterpret_cast<float4*>(out + (size_t)(bt * 16 + ln) * kF + f0) = o;
  }
}

// ---------------------------------------------------------------------------
extern "C" void kernel_launch(void* const* d_in, const int* in_sizes, int n_in,
                              void* d_out, int out_size, void* d_ws, size_t ws_size,
                              hipStream_t stream) {
  (void)in_sizes; (void)n_in; (void)out_size; (void)ws_size;

  const float* x     = (const float*)d_in[0];   // (B, C, H, W) = (B, 64)
  const float* ww    = (const float*)d_in[1];   // (F, P, 64)
  const float* scale = (const float*)d_in[2];   // (1, F)
  const float* bias  = (const float*)d_in[3];   // (1, F)
  float* out = (float*)d_out;                   // (B, F)

  // Workspace: ww_bf16 (F*P*K = 4M elems, 8 MiB) then x_bf16 (B*K, 256 KiB)
  __bf16* wwb = (__bf16*)d_ws;
  __bf16* xbf = wwb + (size_t)kF * kP * kK;

  const int n4_ww = (kF * kP * kK) / 4;
  const int n4_x  = (kB * kK) / 4;
  cvt_f32_to_bf16<<<(n4_ww + 255) / 256, 256, 0, stream>>>(ww, wwb, n4_ww);
  cvt_f32_to_bf16<<<(n4_x  + 255) / 256, 256, 0, stream>>>(x,  xbf, n4_x);

  dim3 grid(kB / 16, kF / 32);   // (128, 32)
  gemm_maxpool_affine<<<grid, 256, 0, stream>>>(xbf, wwb, scale, bias, out);
}
